// CIEGCL_67757404061705
// MI455X (gfx1250) — matrix operand order repaired and checked by reading, not compiled, not run on hardware
//
#include <hip/hip_runtime.h>
#include <hip/hip_bf16.h>

#define NU     100000
#define NI     50000
#define DD     64
#define EDGES  1600000
#define BB     1024
#define INV_T  5.0f      // 1/0.2
#define EPS    1e-8f

typedef __attribute__((ext_vector_type(16))) _Float16 v16h;
typedef __attribute__((ext_vector_type(8)))  _Float16 v8h;
typedef __attribute__((ext_vector_type(8)))  float    v8f;

// ------------------------- utility kernels -------------------------
__global__ void zero_f32(float* __restrict__ p, int n) {
  int i = blockIdx.x * blockDim.x + threadIdx.x;
  if (i < n) p[i] = 0.0f;
}
__global__ void copy_f32(const float* __restrict__ a, float* __restrict__ b, int n) {
  int i = blockIdx.x * blockDim.x + threadIdx.x;
  if (i < n) b[i] = a[i];
}
__global__ void accum_f32(float* __restrict__ s, const float* __restrict__ x, int n) {
  int i = blockIdx.x * blockDim.x + threadIdx.x;
  if (i < n) s[i] += x[i];
}
__global__ void scale_f32(float* __restrict__ p, float f, int n) {
  int i = blockIdx.x * blockDim.x + threadIdx.x;
  if (i < n) p[i] *= f;
}
__global__ void f32_to_f16_k(const float* __restrict__ a, _Float16* __restrict__ b, int n) {
  int i = blockIdx.x * blockDim.x + threadIdx.x;
  if (i < n) b[i] = (_Float16)a[i];
}
__global__ void gather_rows_f16(const float* __restrict__ src, const int* __restrict__ idx,
                                _Float16* __restrict__ dst, int total) {
  int t = blockIdx.x * blockDim.x + threadIdx.x;
  if (t < total) {
    int b = t >> 6, d = t & 63;
    dst[t] = (_Float16)src[idx[b] * DD + d];
  }
}

// ------------------- sample-key sort + edge masking -------------------
__global__ void sort_keys_kernel(const int* __restrict__ users, const int* __restrict__ pos,
                                 long long* __restrict__ skey) {
  __shared__ long long s[BB];
  int t = threadIdx.x;
  s[t] = (long long)users[t] * (long long)NI + (long long)pos[t];
  __syncthreads();
  for (int k = 2; k <= BB; k <<= 1) {
    for (int j = k >> 1; j > 0; j >>= 1) {
      int ixj = t ^ j;
      if (ixj > t) {
        bool up = ((t & k) == 0);
        long long a = s[t], b = s[ixj];
        if ((a > b) == up) { s[t] = b; s[ixj] = a; }
      }
      __syncthreads();
    }
  }
  skey[t] = s[t];
}

__global__ void mark_edges_kernel(const int* __restrict__ rows, const int* __restrict__ cols,
                                  const float* __restrict__ vals,
                                  const long long* __restrict__ skey,
                                  float* __restrict__ ivals, int E) {
  __shared__ long long s[BB];
  for (int i = threadIdx.x; i < BB; i += blockDim.x) s[i] = skey[i];
  __syncthreads();
  int e = blockIdx.x * blockDim.x + threadIdx.x;
  if (e >= E) return;
  long long key = (long long)rows[e] * (long long)NI + (long long)cols[e];
  int lo = 0, hi = BB;                 // lower_bound == searchsorted(left)
  while (lo < hi) { int mid = (lo + hi) >> 1; if (s[mid] < key) lo = mid + 1; else hi = mid; }
  int idx = lo < (BB - 1) ? lo : (BB - 1);
  ivals[e] = (s[idx] == key) ? 0.0f : vals[e];
}

// ------------------- SpMM scatter: dst[di[e]] += v[e]*src[si[e]] -------------------
__global__ void spmm_scatter(const int* __restrict__ dstIdx, const int* __restrict__ srcIdx,
                             const float* __restrict__ vals,
                             const float* __restrict__ src, float* __restrict__ dst, int E) {
  int t = blockIdx.x * blockDim.x + threadIdx.x;
  int e = t >> 4;
  if (e >= E) return;
  float v = vals[e];
  if (v == 0.0f) return;
  int c = (t & 15) << 2;
  const float4 sv = *(const float4*)(src + srcIdx[e] * DD + c);
  float* dp = dst + dstIdx[e] * DD + c;
  atomicAdd(dp + 0, v * sv.x);
  atomicAdd(dp + 1, v * sv.y);
  atomicAdd(dp + 2, v * sv.z);
  atomicAdd(dp + 3, v * sv.w);
}

// ------------- fused GEMM + exp-rowsum via WMMA f16 (f32 accum) -------------
// a_h : gathered batch rows [1024][64] f16 ; emb_h : [N][64] f16
// rowsum[b] += sum_n exp((a_h[b] . emb_h[n]) * INV_T)
// Software-pipelined: B tile for iteration n+1 is loaded before tile n's WMMAs,
// so global_load_b128s overlap the 2 WMMAs + 8 v_exp of the current tile.
__global__ __launch_bounds__(256) void lse_wmma_kernel(
    const _Float16* __restrict__ a_h, const _Float16* __restrict__ emb_h,
    float* __restrict__ rowsum, int ntiles, int nsplit) {
  int lane = threadIdx.x & 31;
  int w = (blockIdx.x * blockDim.x + threadIdx.x) >> 5;
  int bt    = w / nsplit;          // batch tile 0..63
  int split = w % nsplit;          // n-range slice (wave-uniform)
  int h = lane >> 4;               // lane half
  int r = lane & 15;

  // A fragments (lane holds row r; same k order as B): elems 0..15 = K 16h..16h+15 contiguous
  const _Float16* arow = a_h + (bt * 16 + r) * DD;
  v8h lo0 = *(const v8h*)(arow + 16 * h);
  v8h hi0 = *(const v8h*)(arow + 8 + 16 * h);
  v8h lo1 = *(const v8h*)(arow + 32 + 16 * h);
  v8h hi1 = *(const v8h*)(arow + 40 + 16 * h);
  v16h a0, a1;
#pragma unroll
  for (int i = 0; i < 8; i++) { a0[i] = lo0[i]; a0[i + 8] = hi0[i];
                                a1[i] = lo1[i]; a1[i + 8] = hi1[i]; }

  float acc[8] = {0.f, 0.f, 0.f, 0.f, 0.f, 0.f, 0.f, 0.f};
  const float SC = 7.2134752044448170f;   // INV_T * log2(e): exp(x/T) == exp2(x*SC)

  int nt = split;                         // split < nsplit <= ntiles, always valid
  // B fragments (32x16 K-striped): lane holds col r, K = 16h..16h+15 contiguous
  const _Float16* brow = emb_h + (nt * 16 + r) * DD + 16 * h;
  v16h b0 = *(const v16h*)(brow);
  v16h b1 = *(const v16h*)(brow + 32);

  for (int nn = nt + nsplit; nn < ntiles; nn += nsplit) {
    const _Float16* nbrow = emb_h + (nn * 16 + r) * DD + 16 * h;
    v16h nb0 = *(const v16h*)(nbrow);          // prefetch next tile's B (in flight
    v16h nb1 = *(const v16h*)(nbrow + 32);     // during current tile's WMMA+exp)
    int pf = nn + nsplit;                      // L2 prefetch two tiles ahead
    if (pf < ntiles) __builtin_prefetch((const void*)(emb_h + (pf * 16 + r) * DD), 0, 1);
    v8f c = {};
    c = __builtin_amdgcn_wmma_f32_16x16x32_f16(false, a0, false, b0, (short)0, c, false, false);
    c = __builtin_amdgcn_wmma_f32_16x16x32_f16(false, a1, false, b1, (short)0, c, false, false);
#pragma unroll
    for (int v = 0; v < 8; v++) acc[v] += exp2f(c[v] * SC);   // c[v]: M=8h+v, N=r
    b0 = nb0; b1 = nb1;
  }
  {  // drain: last tile
    v8f c = {};
    c = __builtin_amdgcn_wmma_f32_16x16x32_f16(false, a0, false, b0, (short)0, c, false, false);
    c = __builtin_amdgcn_wmma_f32_16x16x32_f16(false, a1, false, b1, (short)0, c, false, false);
#pragma unroll
    for (int v = 0; v < 8; v++) acc[v] += exp2f(c[v] * SC);
  }

#pragma unroll
  for (int v = 0; v < 8; v++) {
    float s = acc[v];
    s += __shfl_xor(s, 1);
    s += __shfl_xor(s, 2);
    s += __shfl_xor(s, 4);
    s += __shfl_xor(s, 8);
    if (r == 0) atomicAdd(&rowsum[bt * 16 + 8 * h + v], s);
  }
}

// ------------------- sampled dot-product terms (BPR + pos score) -------------------
__global__ __launch_bounds__(256) void sampled_loss_kernel(
    const float* __restrict__ ue, const float* __restrict__ ie,
    const float* __restrict__ iue, const float* __restrict__ iie,
    const int* __restrict__ users, const int* __restrict__ pos, const int* __restrict__ neg,
    float* __restrict__ scal) {
  int warp = (blockIdx.x * blockDim.x + threadIdx.x) >> 5;
  int lane = threadIdx.x & 31;
  if (warp >= BB) return;
  int u = users[warp], p = pos[warp], n = neg[warp];
  const float* uerow  = ue  + u * DD;
  const float* perow  = ie  + p * DD;
  const float* nerow  = ie  + n * DD;
  const float* iuerow = iue + u * DD;
  const float* iierow = iie + n * DD;
  float d_un = 0.f, d_up = 0.f, d_pu = 0.f, d_pi = 0.f;
#pragma unroll
  for (int k = 0; k < 2; k++) {
    int d = lane + 32 * k;
    float uv = uerow[d], nv = nerow[d];
    d_un += uv * nv;
    d_up += uv * perow[d];
    d_pu += iuerow[d] * uv;
    d_pi += iierow[d] * nv;
  }
#pragma unroll
  for (int m = 16; m >= 1; m >>= 1) {
    d_un += __shfl_xor(d_un, m);
    d_up += __shfl_xor(d_up, m);
    d_pu += __shfl_xor(d_pu, m);
    d_pi += __shfl_xor(d_pi, m);
  }
  if (lane == 0) {
    float x = d_un - d_up;
    float bpr = fmaxf(x, 0.0f) + log1pf(expf(-fabsf(x)));        // softplus
    float posu = fminf(fmaxf(d_pu * INV_T, -5.0f), 5.0f);
    float posi = fminf(fmaxf(d_pi * INV_T, -5.0f), 5.0f);
    atomicAdd(&scal[0], bpr);
    atomicAdd(&scal[1], posu + posi);
  }
}

// ------------------- finalize: logs, means, scalar combine -------------------
__global__ void finalize_kernel(const float* __restrict__ rowsum_u,
                                const float* __restrict__ rowsum_i,
                                const float* __restrict__ scal, float* __restrict__ out) {
  __shared__ float red[BB];
  int t = threadIdx.x;
  red[t] = logf(rowsum_u[t] + EPS) + logf(rowsum_i[t] + EPS);
  __syncthreads();
  for (int s = BB / 2; s > 0; s >>= 1) {
    if (t < s) red[t] += red[t + s];
    __syncthreads();
  }
  if (t == 0) {
    const float invB = 1.0f / (float)BB;
    float neg_score = red[0] * invB;
    float pos_score = scal[1] * invB;
    float bpr       = scal[0] * invB;
    out[0] = bpr + 0.1f * (neg_score - pos_score);
  }
}

// ------------------------------- driver -------------------------------
extern "C" void kernel_launch(void* const* d_in, const int* in_sizes, int n_in,
                              void* d_out, int out_size, void* d_ws, size_t ws_size,
                              hipStream_t stream) {
  const float* d_uemb = (const float*)d_in[0];
  const float* d_iemb = (const float*)d_in[1];
  const float* d_vals = (const float*)d_in[2];
  const int*   d_rows = (const int*)d_in[3];
  const int*   d_cols = (const int*)d_in[4];
  const int*   d_users = (const int*)d_in[5];
  const int*   d_pos   = (const int*)d_in[6];
  const int*   d_neg   = (const int*)d_in[7];
  float* out = (float*)d_out;

  char* ws = (char*)d_ws;
  size_t off = 0;
  auto take = [&](size_t bytes) -> char* {
    char* p = ws + off;
    off = (off + bytes + 255) & ~(size_t)255;
    return p;
  };
  const size_t UB = (size_t)NU * DD * sizeof(float);
  const size_t IB = (size_t)NI * DD * sizeof(float);
  float* sum_u  = (float*)take(UB);
  float* sum_i  = (float*)take(IB);
  float* isum_u = (float*)take(UB);
  float* isum_i = (float*)take(IB);
  float* cur_u  = (float*)take(UB);
  float* cur_i  = (float*)take(IB);
  float* nxt_u  = (float*)take(UB);
  float* nxt_i  = (float*)take(IB);
  float* ivals  = (float*)take((size_t)EDGES * sizeof(float));
  long long* skey = (long long*)take(BB * sizeof(long long));
  _Float16* ue_h = (_Float16*)take((size_t)NU * DD * sizeof(_Float16));
  _Float16* ie_h = (_Float16*)take((size_t)NI * DD * sizeof(_Float16));
  _Float16* au_h = (_Float16*)take((size_t)BB * DD * sizeof(_Float16));
  _Float16* ai_h = (_Float16*)take((size_t)BB * DD * sizeof(_Float16));
  float* rowsum_u = (float*)take(BB * sizeof(float));
  float* rowsum_i = (float*)take(BB * sizeof(float));
  float* scal     = (float*)take(8 * sizeof(float));
  if (off > ws_size) return;  // workspace too small: bail out safely

  const int T = 256;
  auto G = [](int n, int b) { return (n + b - 1) / b; };
  const int NUE = NU * DD;   // 6.4M
  const int NIE = NI * DD;   // 3.2M

  // 1) sort sampled keys, build intervention edge values
  sort_keys_kernel<<<1, BB, 0, stream>>>(d_users, d_pos, skey);
  mark_edges_kernel<<<G(EDGES, T), T, 0, stream>>>(d_rows, d_cols, d_vals, skey, ivals, EDGES);

  // 2) two LightGCN propagations (layer-averaged)
  auto propagate = [&](const float* vals, float* su, float* si) {
    copy_f32<<<G(NUE, T), T, 0, stream>>>(d_uemb, cur_u, NUE);
    copy_f32<<<G(NIE, T), T, 0, stream>>>(d_iemb, cur_i, NIE);
    copy_f32<<<G(NUE, T), T, 0, stream>>>(d_uemb, su, NUE);
    copy_f32<<<G(NIE, T), T, 0, stream>>>(d_iemb, si, NIE);
    float* cu = cur_u; float* ci = cur_i; float* nu = nxt_u; float* ni = nxt_i;
    for (int layer = 0; layer < 3; layer++) {
      zero_f32<<<G(NUE, T), T, 0, stream>>>(nu, NUE);
      zero_f32<<<G(NIE, T), T, 0, stream>>>(ni, NIE);
      spmm_scatter<<<G(EDGES * 16, T), T, 0, stream>>>(d_rows, d_cols, vals, ci, nu, EDGES);
      spmm_scatter<<<G(EDGES * 16, T), T, 0, stream>>>(d_cols, d_rows, vals, cu, ni, EDGES);
      accum_f32<<<G(NUE, T), T, 0, stream>>>(su, nu, NUE);
      accum_f32<<<G(NIE, T), T, 0, stream>>>(si, ni, NIE);
      float* t0 = cu; cu = nu; nu = t0;
      float* t1 = ci; ci = ni; ni = t1;
    }
    scale_f32<<<G(NUE, T), T, 0, stream>>>(su, 0.25f, NUE);
    scale_f32<<<G(NIE, T), T, 0, stream>>>(si, 0.25f, NIE);
  };
  propagate(d_vals, sum_u, sum_i);   // ue, ie
  propagate(ivals, isum_u, isum_i);  // iue, iie

  // 3) f16 operands for the WMMA logsumexp GEMM
  f32_to_f16_k<<<G(NUE, T), T, 0, stream>>>(sum_u, ue_h, NUE);
  f32_to_f16_k<<<G(NIE, T), T, 0, stream>>>(sum_i, ie_h, NIE);
  gather_rows_f16<<<G(BB * DD, T), T, 0, stream>>>(isum_u, d_users, au_h, BB * DD);
  gather_rows_f16<<<G(BB * DD, T), T, 0, stream>>>(isum_i, d_neg,   ai_h, BB * DD);

  // 4) fused exp-rowsum GEMMs (WMMA)
  zero_f32<<<G(BB, T), T, 0, stream>>>(rowsum_u, BB);
  zero_f32<<<G(BB, T), T, 0, stream>>>(rowsum_i, BB);
  zero_f32<<<1, 8, 0, stream>>>(scal, 8);
  {
    const int nsplit_u = 50;   // 64 tiles * 50 = 3200 waves = 400 blocks
    const int nsplit_i = 24;   // 64 tiles * 24 = 1536 waves = 192 blocks
    lse_wmma_kernel<<<(64 * nsplit_u) / 8, 256, 0, stream>>>(au_h, ue_h, rowsum_u, NU / 16, nsplit_u);
    lse_wmma_kernel<<<(64 * nsplit_i) / 8, 256, 0, stream>>>(ai_h, ie_h, rowsum_i, NI / 16, nsplit_i);
  }

  // 5) sampled BPR + pos-score terms
  sampled_loss_kernel<<<G(BB * 32, T), T, 0, stream>>>(sum_u, sum_i, isum_u, isum_i,
                                                       d_users, d_pos, d_neg, scal);

  // 6) combine to scalar loss
  finalize_kernel<<<1, BB, 0, stream>>>(rowsum_u, rowsum_i, scal, out);
}
